// Attention_88081189306494
// MI455X (gfx1250) — matrix operand order repaired and checked by reading, not compiled, not run on hardware
//
#include <hip/hip_runtime.h>
#include <math.h>

#define H    1024
#define NB   32      // batch
#define L    2048

typedef __bf16 bf16;
typedef __attribute__((ext_vector_type(16))) __bf16 v16bf;
typedef __attribute__((ext_vector_type(8)))  __bf16 v8bf;
typedef __attribute__((ext_vector_type(8)))  float  v8f;

// Fast tanh: gfx1250 has a hardware V_TANH_F32 transcendental.
__device__ __forceinline__ float fast_tanh(float x) {
#if defined(__has_builtin)
#if __has_builtin(__builtin_amdgcn_tanhf)
    return __builtin_amdgcn_tanhf(x);
#else
    float e = __expf(2.f * x);           // v_exp_f32 (TRANS)
    return 1.f - 2.f / (e + 1.f);        // rcp (TRANS) + 2 VALU
#endif
#else
    float e = __expf(2.f * x);
    return 1.f - 2.f / (e + 1.f);
#endif
}

// ---------------------------------------------------------------------------
// K0: convert w_e = attn_w[:, H:2H] (f32) -> bf16, row-major [o][k].
// Row o contiguous over k == per-lane contiguous-K order of the WMMA B frag.
// ---------------------------------------------------------------------------
__global__ __launch_bounds__(256) void k_cvt_we(const float* __restrict__ attn_w,
                                                bf16* __restrict__ wB) {
    int idx = blockIdx.x * 256 + threadIdx.x;       // o*H + k, idx < H*H
    int o = idx >> 10, k = idx & 1023;
    wB[idx] = (bf16)attn_w[(size_t)o * 2048 + 1024 + k];
}

// ---------------------------------------------------------------------------
// K1: h_proj[b,o] = sum_h hidden[b,h] * attn_w[o, h] + attn_b[o]   (f32)
// ---------------------------------------------------------------------------
__global__ __launch_bounds__(256) void k_hproj(const float* __restrict__ hidden,
                                               const float* __restrict__ attn_w,
                                               const float* __restrict__ attn_b,
                                               float* __restrict__ h_proj) {
    int idx = blockIdx.x * 256 + threadIdx.x;       // b*H + o
    int b = idx >> 10, o = idx & 1023;
    const float4* hv4 = (const float4*)(hidden + (size_t)b * H);
    const float4* wr4 = (const float4*)(attn_w + (size_t)o * 2048);  // w_h row o
    float acc = attn_b[o];
#pragma unroll 4
    for (int h = 0; h < H / 4; ++h) {
        float4 a = hv4[h], w = wr4[h];
        acc += a.x * w.x + a.y * w.y + a.z * w.z + a.w * w.w;
    }
    h_proj[idx] = acc;
}

// ---------------------------------------------------------------------------
// K2: fused  e_proj -> tanh -> dot(v_w) -> scores[b,l]
// Block: one batch b, one 32-row l-tile (two M-tiles). 8 waves split N=1024.
// K is processed in two staged halves of 512 so LDS stays ~33 KB while each
// B fragment (L2) feeds TWO wmma ops (M-tile 0 and 1) -> half the L2 traffic.
// Accumulators (8 nt x 2 mt x v8f) persist across the K halves.
// ---------------------------------------------------------------------------
#define KH   512
#define LDP  520   // LDS row pitch in bf16: 1040 B -> 4-bank rotation per row

__global__ __launch_bounds__(256) void k_scores(const float* __restrict__ enc,
                                                const bf16*  __restrict__ wB,
                                                const float* __restrict__ h_proj,
                                                const float* __restrict__ v_w,
                                                float* __restrict__ scores) {
    __shared__ bf16  sA[32 * LDP];      // ~33 KB
    __shared__ float sS[32 * 8];        // per-row, per-wave partial scores

    const int b  = blockIdx.x >> 6;             // 64 l-tiles (of 32 rows) per batch
    const int l0 = (blockIdx.x & 63) << 5;
    const size_t encBase = ((size_t)b * L + l0) * H;

    const int wave  = threadIdx.x >> 5;
    const int lane  = threadIdx.x & 31;
    const int laneN = lane & 15;
    const int hi    = lane >> 4;        // half-wave select
    const int akb   = hi ? 16 : 0;      // A frag K-split: lo lanes 0..15, hi 16..31

    v8f acc[8][2];
#pragma unroll
    for (int nt = 0; nt < 8; ++nt) { acc[nt][0] = {}; acc[nt][1] = {}; }

    for (int kh = 0; kh < 2; ++kh) {
        const int kbase = kh * KH;
        // Stage 32 rows x 512 K as bf16: 8 f32 -> 8 bf16 -> one 16B DS store.
        for (int j = 0; j < 8; ++j) {
            int chunk = threadIdx.x + j * 256;        // 2048 chunks of 8 elems
            int r = chunk >> 6;                       // 64 chunks per row
            int c = (chunk & 63) << 3;
            const float* gp = enc + encBase + (size_t)r * H + kbase + c;
            float4 f0 = *(const float4*)(gp);
            float4 f1 = *(const float4*)(gp + 4);
            v8bf o;
            o[0] = (bf16)f0.x; o[1] = (bf16)f0.y; o[2] = (bf16)f0.z; o[3] = (bf16)f0.w;
            o[4] = (bf16)f1.x; o[5] = (bf16)f1.y; o[6] = (bf16)f1.z; o[7] = (bf16)f1.w;
            *(v8bf*)(sA + r * LDP + c) = o;
        }
        __syncthreads();

        const bf16* arow0 = sA + laneN * LDP;          // M-tile 0: rows 0..15
        const bf16* arow1 = sA + (laneN + 16) * LDP;   // M-tile 1: rows 16..31

#pragma unroll
        for (int nt = 0; nt < 8; ++nt) {
            const int n = wave * 128 + nt * 16 + laneN;
            const bf16* bcol = wB + (size_t)n * 1024 + kbase;
            v8f d0 = acc[nt][0];
            v8f d1 = acc[nt][1];
#pragma unroll
            for (int k0 = 0; k0 < KH; k0 += 32) {
                // B fragment (32x16 bf16): lanes 0-15 K=0..15, lanes 16-31 K=16..31
                v8bf b0 = *(const v8bf*)(bcol + k0 + hi * 16);
                v8bf b1 = *(const v8bf*)(bcol + k0 + hi * 16 + 8);
                v16bf Bm = __builtin_shufflevector(b0, b1,
                            0,1,2,3,4,5,6,7,8,9,10,11,12,13,14,15);
                // A fragments for both M-tiles (two 16B LDS loads each)
                v8bf a0 = *(const v8bf*)(arow0 + k0 + akb);
                v8bf a1 = *(const v8bf*)(arow0 + k0 + akb + 8);
                v16bf A0 = __builtin_shufflevector(a0, a1,
                            0,1,2,3,4,5,6,7,8,9,10,11,12,13,14,15);
                v8bf c0 = *(const v8bf*)(arow1 + k0 + akb);
                v8bf c1 = *(const v8bf*)(arow1 + k0 + akb + 8);
                v16bf A1 = __builtin_shufflevector(c0, c1,
                            0,1,2,3,4,5,6,7,8,9,10,11,12,13,14,15);
                d0 = __builtin_amdgcn_wmma_f32_16x16x32_bf16(
                        false, A0, false, Bm, (short)0, d0, false, false);
                d1 = __builtin_amdgcn_wmma_f32_16x16x32_bf16(
                        false, A1, false, Bm, (short)0, d1, false, false);
            }
            acc[nt][0] = d0;
            acc[nt][1] = d1;
        }
        __syncthreads();   // all waves done reading sA before restage
    }

    // Fold energy -> per-lane score partials.  D layout: lane = column n,
    // VGPR r = row m = r + 8*hi (+16 for M-tile 1).
    float racc[2][8];
#pragma unroll
    for (int mt = 0; mt < 2; ++mt)
#pragma unroll
        for (int r = 0; r < 8; ++r) racc[mt][r] = 0.f;

#pragma unroll
    for (int nt = 0; nt < 8; ++nt) {
        const int n = wave * 128 + nt * 16 + laneN;
        const float hp = h_proj[b * H + n];
        const float vw = v_w[n];
#pragma unroll
        for (int r = 0; r < 8; ++r) {
            racc[0][r] += fast_tanh(acc[nt][0][r] + hp) * vw;
            racc[1][r] += fast_tanh(acc[nt][1][r] + hp) * vw;
        }
    }

    // Reduce over the 16 n-lanes of each half-wave.
#pragma unroll
    for (int mt = 0; mt < 2; ++mt)
#pragma unroll
        for (int r = 0; r < 8; ++r) {
            float v = racc[mt][r];
            v += __shfl_xor(v, 1);
            v += __shfl_xor(v, 2);
            v += __shfl_xor(v, 4);
            v += __shfl_xor(v, 8);
            if (laneN == 0) sS[(mt * 16 + r + 8 * hi) * 8 + wave] = v;
        }
    __syncthreads();
    if (threadIdx.x < 32) {
        float s = 0.f;
#pragma unroll
        for (int w = 0; w < 8; ++w) s += sS[threadIdx.x * 8 + w];
        scores[(size_t)b * L + l0 + threadIdx.x] = s;
    }
}

// ---------------------------------------------------------------------------
// K3: softmax over L per batch, in-place on scores.
// ---------------------------------------------------------------------------
__global__ __launch_bounds__(256) void k_softmax(float* __restrict__ scores) {
    __shared__ float red[256];
    float* row = scores + (size_t)blockIdx.x * L;
    float e[8];
    float m = -1e30f;
#pragma unroll
    for (int i = 0; i < 8; ++i) { e[i] = row[threadIdx.x + i * 256]; m = fmaxf(m, e[i]); }
    red[threadIdx.x] = m; __syncthreads();
    for (int s = 128; s > 0; s >>= 1) {
        if (threadIdx.x < s) red[threadIdx.x] = fmaxf(red[threadIdx.x], red[threadIdx.x + s]);
        __syncthreads();
    }
    m = red[0]; __syncthreads();
    float sum = 0.f;
#pragma unroll
    for (int i = 0; i < 8; ++i) { e[i] = __expf(e[i] - m); sum += e[i]; }
    red[threadIdx.x] = sum; __syncthreads();
    for (int s = 128; s > 0; s >>= 1) {
        if (threadIdx.x < s) red[threadIdx.x] += red[threadIdx.x + s];
        __syncthreads();
    }
    float inv = 1.f / red[0];
#pragma unroll
    for (int i = 0; i < 8; ++i) row[threadIdx.x + i * 256] = e[i] * inv;
}

// ---------------------------------------------------------------------------
// K4a: partial context over an L-chunk of 256:  part[b][c][h]
// 256 blocks in flight to saturate HBM on the 256 MB encoder stream.
// Deterministic (no float atomics).
// ---------------------------------------------------------------------------
__global__ __launch_bounds__(256) void k_ctx_part(const float* __restrict__ enc,
                                                  const float* __restrict__ weights,
                                                  float* __restrict__ part) {
    __shared__ float w[256];
    const int b = blockIdx.x >> 3;
    const int c = blockIdx.x & 7;
    const int lbase = c * 256;
    w[threadIdx.x] = weights[(size_t)b * L + lbase + threadIdx.x];
    __syncthreads();
    const int h = threadIdx.x * 4;
    const float* base = enc + ((size_t)b * L + lbase) * H + h;
    float4 acc = make_float4(0.f, 0.f, 0.f, 0.f);
    for (int l = 0; l < 256; ++l) {
        float4 e4 = *(const float4*)(base + (size_t)l * H);
        float wl = w[l];
        acc.x += e4.x * wl; acc.y += e4.y * wl;
        acc.z += e4.z * wl; acc.w += e4.w * wl;
    }
    *(float4*)(part + ((size_t)(b * 8 + c)) * H + h) = acc;
}

// K4b: out[b,h] = sum_c part[b][c][h]
__global__ __launch_bounds__(256) void k_ctx_reduce(const float* __restrict__ part,
                                                    float* __restrict__ out) {
    int idx = blockIdx.x * 256 + threadIdx.x;   // b*H + h
    int b = idx >> 10, h = idx & 1023;
    float s = 0.f;
#pragma unroll
    for (int c = 0; c < 8; ++c) s += part[((size_t)(b * 8 + c)) * H + h];
    out[idx] = s;
}

// ---------------------------------------------------------------------------
extern "C" void kernel_launch(void* const* d_in, const int* in_sizes, int n_in,
                              void* d_out, int out_size, void* d_ws, size_t ws_size,
                              hipStream_t stream) {
    const float* hidden = (const float*)d_in[0];   // (B,H)
    const float* enc    = (const float*)d_in[1];   // (B,L,H)
    const float* attn_w = (const float*)d_in[2];   // (H,2H)
    const float* attn_b = (const float*)d_in[3];   // (H,)
    const float* v_w    = (const float*)d_in[4];   // (H,)
    float* out = (float*)d_out;                    // (B,H)

    // Workspace layout (bytes):
    //   [0x000000, 0x200000)  w_e bf16 (2 MB)
    //   [0x200000, 0x220000)  h_proj f32 (128 KB)
    //   [0x220000, 0x260000)  scores/weights f32 (256 KB)
    //   [0x260000, 0x360000)  context partials f32 (1 MB)
    bf16*  wB     = (bf16*)d_ws;
    float* h_proj = (float*)((char*)d_ws + 0x200000u);
    float* scores = (float*)((char*)d_ws + 0x220000u);
    float* part   = (float*)((char*)d_ws + 0x260000u);

    k_cvt_we    <<<dim3((H * H) / 256), dim3(256), 0, stream>>>(attn_w, wB);
    k_hproj     <<<dim3((NB * H) / 256), dim3(256), 0, stream>>>(hidden, attn_w, attn_b, h_proj);
    k_scores    <<<dim3(NB * (L / 32)), dim3(256), 0, stream>>>(enc, wB, h_proj, v_w, scores);
    k_softmax   <<<dim3(NB), dim3(256), 0, stream>>>(scores);
    k_ctx_part  <<<dim3(NB * 8), dim3(256), 0, stream>>>(enc, scores, part);
    k_ctx_reduce<<<dim3((NB * H) / 256), dim3(256), 0, stream>>>(part, out);
}
